// f_phi_78812649881983
// MI455X (gfx1250) — compile-verified, matrix operand order checked by reading, not run on hardware
//
#include <hip/hip_runtime.h>

// ---------------------------------------------------------------------------
// GCNConv + grouped L2 norm for MI455X (gfx1250, wave32, WMMA).
//
//   h = D^-1/2 (A+I) D^-1/2 (x W) + b_gcn ;  out[g,i] = ||h[i,32g:32g+32]|| + bias[g]
//
// Pass 0 streams adj once (NT loads): emits deg^-1/2 AND ahat_f16 = (A+I) as
// exact 0/1 f16 (32 MB, L2-resident). Pass 1: y = D^-1/2 (xW) in f16, stored
// column-major. Pass 2: 34.4 GFLOP GEMM on v_wmma_f32_16x16x32_f16 with f32
// accum + fused grouped-L2 epilogue (h never touches memory). K staged in
// 64-wide chunks (16 WMMAs per barrier), double-buffered LDS (pitch 9 uint4
// -> all fragment loads are aligned, bank-conflict-free ds_load_b128),
// staging via GLOBAL_LOAD_ASYNC_TO_LDS_B128 (ASYNCcnt).
// ---------------------------------------------------------------------------

typedef __attribute__((ext_vector_type(16))) _Float16 v16h;
typedef __attribute__((ext_vector_type(8)))  float    v8f;
typedef __attribute__((ext_vector_type(4)))  float    v4f;
typedef __attribute__((ext_vector_type(4)))  int      v4i;

#define NN   4096   // nodes
#define INC  32     // in channels
#define OUTC 1024   // out channels

#if defined(__has_builtin)
#if __has_builtin(__builtin_amdgcn_global_load_async_to_lds_b128)
#define USE_ASYNC_LDS 1
#endif
#endif

#ifdef USE_ASYNC_LDS
typedef __attribute__((address_space(1))) v4i* gv4p;   // global int4*
typedef __attribute__((address_space(3))) v4i* lv4p;   // LDS int4*
__device__ __forceinline__ void async_cp16(const void* g, void* l) {
    // per-lane 16B global->LDS copy, tracked by ASYNCcnt
    __builtin_amdgcn_global_load_async_to_lds_b128(
        (gv4p)(uintptr_t)g, (lv4p)(unsigned)(uintptr_t)l, 0, 0);
}
__device__ __forceinline__ void async_wait0() {
#if __has_builtin(__builtin_amdgcn_s_wait_asynccnt)
    __builtin_amdgcn_s_wait_asynccnt(0);
#else
    asm volatile("s_wait_asynccnt 0" ::: "memory");
#endif
}
#endif

// CDNA5 16-bit A/B fragment K-index (ISA 7.12.2):
// lanes 0-15: K = 0..7 (v0..v3), 16..23 (v4..v7); lanes 16-31: 8..15, 24..31.
__device__ __forceinline__ int kmap(int i, int lane) {
    int kb = (lane >> 4) << 3;
    return kb + i + ((i >= 8) ? 8 : 0);
}

union U16 { uint4 q[2]; v16h v; };          // fragment = 2x b128
union U8h { uint4 q; _Float16 h[8]; };      // 8 halves = 1x b128

// ---------------------------------------------------------------------------
// K0: single pass over adj (64 MB, NT): dis[i] = rsqrt(1+nnz(row i)),
//     ahat[i][j] = (adj[i][j]!=0 || i==j) ? 1.0h : 0.0h   (exact in f16)
// ---------------------------------------------------------------------------
__global__ __launch_bounds__(256) void deg_kernel(const float* __restrict__ adj,
                                                  float* __restrict__ dis,
                                                  _Float16* __restrict__ ahat) {
    const int wave = threadIdx.x >> 5;
    const int lane = threadIdx.x & 31;
    const int row  = blockIdx.x * 8 + wave;
    const v4f* rp = (const v4f*)(adj + (size_t)row * NN);
    unsigned* hp = (unsigned*)(ahat + (size_t)row * NN);
    int cnt = 0;
    #pragma unroll 4
    for (int it = 0; it < NN / (32 * 4); ++it) {
        v4f v = __builtin_nontemporal_load(&rp[it * 32 + lane]);
        cnt += (v.x != 0.f) + (v.y != 0.f) + (v.z != 0.f) + (v.w != 0.f);
        const int j = it * 128 + lane * 4;
        unsigned h0 = ((v.x != 0.f) | (row == j + 0)) ? 0x3C00u : 0u;
        unsigned h1 = ((v.y != 0.f) | (row == j + 1)) ? 0x3C00u : 0u;
        unsigned h2 = ((v.z != 0.f) | (row == j + 2)) ? 0x3C00u : 0u;
        unsigned h3 = ((v.w != 0.f) | (row == j + 3)) ? 0x3C00u : 0u;
        hp[it * 64 + lane * 2 + 0] = h0 | (h1 << 16);
        hp[it * 64 + lane * 2 + 1] = h2 | (h3 << 16);
    }
    #pragma unroll
    for (int off = 16; off > 0; off >>= 1) cnt += __shfl_xor(cnt, off, 32);
    if (lane == 0) dis[row] = rsqrtf((float)(cnt + 1));
}

// ---------------------------------------------------------------------------
// K1: y[n, m] = (f16)( dis[m] * sum_k x[m,k] * W[k,n] )   (column-major)
// ---------------------------------------------------------------------------
__global__ __launch_bounds__(256) void xw_kernel(const float* __restrict__ x,
                                                 const float* __restrict__ W,
                                                 const float* __restrict__ dis,
                                                 _Float16* __restrict__ y) {
    const int w    = threadIdx.x >> 5;
    const int lane = threadIdx.x & 31;
    const int l16  = lane & 15;
    const int row0 = (blockIdx.x >> 3) * 16;
    const int n0   = (blockIdx.x & 7) * 128 + w * 16;
    const int kb   = (lane >> 4) << 3;

    const int arow = row0 + l16;
    const float4* xp = (const float4*)(x + arow * INC + kb);
    float4 x0 = xp[0], x1 = xp[1];
    const float4* xq = (const float4*)(x + arow * INC + kb + 16);
    float4 x2 = xq[0], x3 = xq[1];
    v16h a;
    a[0]=(_Float16)x0.x; a[1]=(_Float16)x0.y; a[2]=(_Float16)x0.z; a[3]=(_Float16)x0.w;
    a[4]=(_Float16)x1.x; a[5]=(_Float16)x1.y; a[6]=(_Float16)x1.z; a[7]=(_Float16)x1.w;
    a[8]=(_Float16)x2.x; a[9]=(_Float16)x2.y; a[10]=(_Float16)x2.z; a[11]=(_Float16)x2.w;
    a[12]=(_Float16)x3.x; a[13]=(_Float16)x3.y; a[14]=(_Float16)x3.z; a[15]=(_Float16)x3.w;

    const int bcol = n0 + l16;
    v16h b;
    #pragma unroll
    for (int i = 0; i < 16; ++i) b[i] = (_Float16)W[kmap(i, lane) * OUTC + bcol];

    v8f c = {};
    c = __builtin_amdgcn_wmma_f32_16x16x32_f16(false, a, false, b,
                                               (short)0, c, false, false);

    const int m0 = row0 + ((lane >> 4) << 3);
    const float4* dp = (const float4*)(dis + m0);
    float4 d0 = dp[0], d1 = dp[1];
    U8h uy;
    uy.h[0] = (_Float16)(d0.x * c[0]); uy.h[1] = (_Float16)(d0.y * c[1]);
    uy.h[2] = (_Float16)(d0.z * c[2]); uy.h[3] = (_Float16)(d0.w * c[3]);
    uy.h[4] = (_Float16)(d1.x * c[4]); uy.h[5] = (_Float16)(d1.y * c[5]);
    uy.h[6] = (_Float16)(d1.z * c[6]); uy.h[7] = (_Float16)(d1.w * c[7]);
    *(uint4*)(y + (size_t)bcol * NN + m0) = uy.q;
}

// ---------------------------------------------------------------------------
// K2: acc = ahat_f16 @ y_f16 (f32 accum) + fused grouped-L2 epilogue.
// 8 waves / 128x128 tile; wave = 64 rows x 32 cols (one channel group).
// K staged 64 at a time: 16 WMMAs per wave per barrier.
// ---------------------------------------------------------------------------
__global__ __launch_bounds__(256) void agg_kernel(const _Float16* __restrict__ ahat,
                                                  const _Float16* __restrict__ y,
                                                  const float* __restrict__ dis,
                                                  const float* __restrict__ b_gcn,
                                                  const float* __restrict__ bias,
                                                  float* __restrict__ out) {
    __shared__ uint4 AshQ[2][128 * 9];   // [m][K: 8 uint4 data + 1 pad]
    __shared__ uint4 BshQ[2][128 * 9];   // [n][K: same]

    const int tid  = threadIdx.x;
    const int lane = tid & 31;
    const int w    = tid >> 5;
    const int l16  = lane & 15;
    const int r0   = blockIdx.y * 128;
    const int c0   = blockIdx.x * 128;
    const int wr0  = (w >> 2) * 64;
    const int wc0  = (w & 3) * 32;

    const int ar  = tid >> 1;            // A staging: row 0..127
    const int ah  = tid & 1;             // A: which 32-K half
    const int bn  = tid & 127;           // B staging: col 0..127
    const int bkh = tid >> 7;            // B: which 32-K half
    const int gr  = r0 + ar;

    v8f acc[4][2] = {};

#ifdef USE_ASYNC_LDS
    auto stage_async = [&](int kk, int b) {
        const _Float16* ga = ahat + (size_t)gr * NN + kk + ah * 32;
        uint4* la = &AshQ[b][ar * 9 + ah * 4];
        async_cp16(ga,      la);     async_cp16(ga + 8,  la + 1);
        async_cp16(ga + 16, la + 2); async_cp16(ga + 24, la + 3);
        const _Float16* gb = y + (size_t)(c0 + bn) * NN + kk + bkh * 32;
        uint4* lb = &BshQ[b][bn * 9 + bkh * 4];
        async_cp16(gb,      lb);     async_cp16(gb + 8,  lb + 1);
        async_cp16(gb + 16, lb + 2); async_cp16(gb + 24, lb + 3);
    };
    stage_async(0, 0);
    async_wait0();
#else
    uint4 qa[4], qb[4];
    auto load_tile = [&](int kk) {
        const uint4* ap = (const uint4*)(ahat + (size_t)gr * NN + kk + ah * 32);
        qa[0] = ap[0]; qa[1] = ap[1]; qa[2] = ap[2]; qa[3] = ap[3];
        if (kk + 64 < NN)
            __builtin_prefetch(ahat + (size_t)gr * NN + kk + 64 + ah * 32, 0, 0);
        const uint4* yp = (const uint4*)(y + (size_t)(c0 + bn) * NN + kk + bkh * 32);
        qb[0] = yp[0]; qb[1] = yp[1]; qb[2] = yp[2]; qb[3] = yp[3];
    };
    auto store_tile = [&](int b) {
        uint4* adst = &AshQ[b][ar * 9 + ah * 4];
        adst[0] = qa[0]; adst[1] = qa[1]; adst[2] = qa[2]; adst[3] = qa[3];
        uint4* bdst = &BshQ[b][bn * 9 + bkh * 4];
        bdst[0] = qb[0]; bdst[1] = qb[1]; bdst[2] = qb[2]; bdst[3] = qb[3];
    };
    load_tile(0);
    store_tile(0);
#endif
    __syncthreads();

    const int kq = lane >> 4;            // uint4 offset 0/1 within 32-K chunk
    int buf = 0;
    for (int kk = 0; kk < NN; kk += 64) {
        const bool more = (kk + 64) < NN;
#ifdef USE_ASYNC_LDS
        if (more) stage_async(kk + 64, buf ^ 1);
#else
        if (more) load_tile(kk + 64);
#endif
        #pragma unroll
        for (int s = 0; s < 2; ++s) {        // two 32-K sub-chunks
            const int sq = s * 4 + kq;
            U16 bf0, bf1;
            const int nf = wc0 + l16;
            bf0.q[0] = BshQ[buf][nf * 9 + sq];        bf0.q[1] = BshQ[buf][nf * 9 + sq + 2];
            bf1.q[0] = BshQ[buf][(nf + 16) * 9 + sq]; bf1.q[1] = BshQ[buf][(nf + 16) * 9 + sq + 2];
            #pragma unroll
            for (int rt = 0; rt < 4; ++rt) {
                U16 af;
                const int m = wr0 + rt * 16 + l16;
                af.q[0] = AshQ[buf][m * 9 + sq];
                af.q[1] = AshQ[buf][m * 9 + sq + 2];
                acc[rt][0] = __builtin_amdgcn_wmma_f32_16x16x32_f16(
                    false, af.v, false, bf0.v, (short)0, acc[rt][0], false, false);
                acc[rt][1] = __builtin_amdgcn_wmma_f32_16x16x32_f16(
                    false, af.v, false, bf1.v, (short)0, acc[rt][1], false, false);
            }
        }
#ifdef USE_ASYNC_LDS
        if (more) async_wait0();
#else
        if (more) store_tile(buf ^ 1);
#endif
        __syncthreads();
        if (more) buf ^= 1;
    }

    // ---- fused epilogue: wave's 32 cols == one channel group
    const float bg0 = b_gcn[c0 + wc0 + l16];
    const float bg1 = b_gcn[c0 + wc0 + 16 + l16];
    const int   g   = (c0 + wc0) >> 5;
    const float bs  = bias[g];
    const int   mo  = (lane >> 4) << 3;
    #pragma unroll
    for (int rt = 0; rt < 4; ++rt) {
        #pragma unroll
        for (int r = 0; r < 8; ++r) {
            int row = r0 + wr0 + rt * 16 + mo + r;
            float d  = dis[row];
            float h0 = d * acc[rt][0][r] + bg0;
            float h1 = d * acc[rt][1][r] + bg1;
            float s  = h0 * h0 + h1 * h1;
            #pragma unroll
            for (int off = 1; off < 16; off <<= 1) s += __shfl_xor(s, off, 32);
            if (l16 == 0) out[g * NN + row] = sqrtf(s) + bs;
        }
    }
}

// ---------------------------------------------------------------------------
extern "C" void kernel_launch(void* const* d_in, const int* in_sizes, int n_in,
                              void* d_out, int out_size, void* d_ws, size_t ws_size,
                              hipStream_t stream) {
    const float* x     = (const float*)d_in[0];  // [1,4096,32]
    const float* adj   = (const float*)d_in[1];  // [4096,4096]
    const float* W     = (const float*)d_in[2];  // [32,1024]
    const float* b_gcn = (const float*)d_in[3];  // [1024]
    const float* bias  = (const float*)d_in[4];  // [32]
    float* out = (float*)d_out;                  // [1,32,4096]

    float*    dis  = (float*)d_ws;                                       // 16 KB
    _Float16* y    = (_Float16*)((char*)d_ws + NN * sizeof(float));      // 8 MB
    _Float16* ahat = (_Float16*)((char*)d_ws + NN * sizeof(float)
                                 + (size_t)OUTC * NN * sizeof(_Float16)); // 32 MB

    deg_kernel<<<NN / 8, 256, 0, stream>>>(adj, dis, ahat);
    xw_kernel<<<(NN / 16) * 8, 256, 0, stream>>>(x, W, dis, y);
    dim3 g2(OUTC / 128, NN / 128);
    agg_kernel<<<g2, 256, 0, stream>>>(ahat, y, dis, b_gcn, bias, out);
}
